// Critic_40819369181496
// MI455X (gfx1250) — matrix operand order repaired and checked by reading, not compiled, not run on hardware
//
#include <hip/hip_runtime.h>
#include <hip/hip_bf16.h>
#include <math.h>

// ---------------------------------------------------------------------------
// MI455X strategy: ~1 MFLOP total, ~280KB weights -> pure latency problem.
// One kernel; block b = branch b (2 blocks x 256 threads = 8 waves), all
// intermediates in LDS (~27KB << 320KB/WGP).  The only GEMM-shaped op -- the
// LSTM input projection for all 4 timesteps, X(4x192) @ Wih^T(192x128) -- is
// hoisted out of the sequential scan and done with v_wmma_f32_16x16x32_f16:
// M padded 4->16 via a zero-filled LDS panel (branch-free A loads), 8 N-tiles
// (one per wave), 6 K-tiles each.  Fragments are built with paired (float2)
// loads since the ISA layout puts two contiguous K values per VGPR.
// ---------------------------------------------------------------------------

typedef __attribute__((ext_vector_type(16))) _Float16 v16h;
typedef __attribute__((ext_vector_type(8)))  float    v8f;

union V16H { v16h v; _Float16 h[16]; };
union V8F  { v8f  v; float    f[8];  };

struct KArgs {
    const float* in[38];
    float*       out;
};

__device__ __forceinline__ float sigm(float x) { return 1.0f / (1.0f + __expf(-x)); }

__global__ __launch_bounds__(256) void critic_branch_kernel(KArgs A) {
    const int br  = blockIdx.x;            // branch 0 / 1
    const int tid = threadIdx.x;

    const float* cv   = A.in[0];           // (4,10,20)
    const float* num  = A.in[1];           // (1,6)
    const float* h0   = A.in[2 + 2 * br];
    const float* c0   = A.in[3 + 2 * br];
    const float* const* P = A.in + 6 + 16 * br;
    const float* w1  = P[0];  const float* b1  = P[1];
    const float* w2  = P[2];  const float* b2  = P[3];
    const float* w3  = P[4];  const float* b3  = P[5];
    const float* Wih = P[6];  const float* Whh = P[7];
    const float* bih = P[8];  const float* bhh = P[9];
    const float* wn1 = P[10]; const float* bn1 = P[11];
    const float* wn2 = P[12]; const float* bn2 = P[13];
    const float* wo  = P[14]; const float* bo  = P[15];

    __shared__ float s_cv[800];            // (4,10,20)
    __shared__ float s_x1[800];            // conv1 out + residual
    __shared__ float s_x2[1152];           // (8,8,18)
    __shared__ float s_x3[16 * 192];       // A panel: rows 0..3 real, 4..15 zero
    __shared__ float s_xwih[4 * 128];      // WMMA result: X @ Wih^T
    __shared__ float s_g[128];             // LSTM gates for one step
    __shared__ float s_h[32], s_c[32];
    __shared__ float s_ys[128];            // h_t for t=0..3
    __shared__ float s_y1[64], s_y2[64];   // num-MLP

    // Stream Wih (98KB) toward the caches while convs run out of LDS.
    for (int off = tid * 32; off < 128 * 192; off += 256 * 32)
        __builtin_prefetch(Wih + off, 0, 3);

    // ---- stage input + zero-fill the padded A-panel rows ------------------
    for (int i = tid; i < 800; i += 256) s_cv[i] = cv[i];
    for (int i = 768 + tid; i < 16 * 192; i += 256) s_x3[i] = 0.0f;
    __syncthreads();

    // ---- conv1 4->4, 3x3, pad 1, relu, residual ---------------------------
    for (int idx = tid; idx < 800; idx += 256) {
        const int co = idx / 200, rem = idx % 200, y = rem / 20, x = rem % 20;
        float acc = b1[co];
#pragma unroll
        for (int ci = 0; ci < 4; ++ci)
#pragma unroll
            for (int ky = 0; ky < 3; ++ky) {
                const int iy = y + ky - 1;
                if (iy < 0 || iy >= 10) continue;
#pragma unroll
                for (int kx = 0; kx < 3; ++kx) {
                    const int ix = x + kx - 1;
                    if (ix < 0 || ix >= 20) continue;
                    acc += s_cv[ci * 200 + iy * 20 + ix] *
                           w1[co * 36 + ci * 9 + ky * 3 + kx];
                }
            }
        s_x1[idx] = s_cv[idx] + fmaxf(acc, 0.0f);
    }
    __syncthreads();

    // ---- conv2 4->8, 3x3, pad 0, relu : out (8,8,18) ----------------------
    for (int idx = tid; idx < 1152; idx += 256) {
        const int co = idx / 144, rem = idx % 144, y = rem / 18, x = rem % 18;
        float acc = b2[co];
#pragma unroll
        for (int ci = 0; ci < 4; ++ci)
#pragma unroll
            for (int ky = 0; ky < 3; ++ky)
#pragma unroll
                for (int kx = 0; kx < 3; ++kx)
                    acc += s_x1[ci * 200 + (y + ky) * 20 + (x + kx)] *
                           w2[co * 36 + ci * 9 + ky * 3 + kx];
        s_x2[idx] = fmaxf(acc, 0.0f);
    }
    __syncthreads();

    // ---- conv3 8->8, 3x3, pad 0, relu : out (8,6,16) ----------------------
    for (int idx = tid; idx < 768; idx += 256) {
        const int co = idx / 96, rem = idx % 96, y = rem / 16, x = rem % 16;
        float acc = b3[co];
#pragma unroll
        for (int ci = 0; ci < 8; ++ci)
#pragma unroll
            for (int ky = 0; ky < 3; ++ky)
#pragma unroll
                for (int kx = 0; kx < 3; ++kx)
                    acc += s_x2[ci * 144 + (y + ky) * 18 + (x + kx)] *
                           w3[co * 72 + ci * 9 + ky * 3 + kx];
        s_x3[idx] = fmaxf(acc, 0.0f);   // flat == (4 timesteps, 192 features)
    }
    __syncthreads();

    // ---- WMMA: s_xwih(4x128) = X(16x192 padded) @ Wih^T(192x128) ----------
    // wave w owns N columns [16w,16w+16); 6 K-tiles of 32; f32 accumulate.
    // A layout (ISA 7.12.2): lane = M%16 (+ K-half by lane/16), halves hold
    // contiguous K pairs -> build fragments from float2 loads, branch-free.
    {
        const int wv   = tid >> 5;
        const int lane = tid & 31;
        const int m    = lane & 15;
        const int g    = lane >> 4;

        V8F acc; acc.v = (v8f){};
#pragma unroll
        for (int kt = 0; kt < 6; ++kt) {
            V16H a, b;
            const float2* xrow =
                (const float2*)(s_x3 + m * 192 + kt * 32 + 8 * g);
#pragma unroll
            for (int p = 0; p < 4; ++p) {
                const float2 lo = xrow[p];        // K = 8g + 2p, 2p+1
                a.h[2 * p]     = (_Float16)lo.x;
                a.h[2 * p + 1] = (_Float16)lo.y;
                const float2 hi = xrow[8 + p];    // K = 8g + 16 + 2p, +1
                a.h[8 + 2 * p]     = (_Float16)hi.x;
                a.h[8 + 2 * p + 1] = (_Float16)hi.y;
            }
            const float2* wrow =
                (const float2*)(Wih + (wv * 16 + m) * 192 + kt * 32 + 8 * g);
#pragma unroll
            for (int p = 0; p < 8; ++p) {
                const float2 w = wrow[p + (p >= 4 ? 4 : 0)]; // K = 8g + 2p (+8 if p>=4), +1
                b.h[2 * p]     = (_Float16)w.x;
                b.h[2 * p + 1] = (_Float16)w.y;
            }
            acc.v = __builtin_amdgcn_wmma_f32_16x16x32_f16(
                false, a.v, false, b.v, (short)0, acc.v, false, false);
        }
        // D: M = r + 8*g, N = wv*16 + m ; only rows 0..3 are real timesteps.
        if (g == 0) {
#pragma unroll
            for (int r = 0; r < 4; ++r)
                s_xwih[r * 128 + wv * 16 + m] = acc.f[r];
        }
    }

    // ---- LSTM scan, T=4 ----------------------------------------------------
    if (tid < 32) { s_h[tid] = h0[tid]; s_c[tid] = c0[tid]; }
    __syncthreads();

    for (int t = 0; t < 4; ++t) {
        if (tid < 128) {
            float g = s_xwih[t * 128 + tid] + bih[tid] + bhh[tid];
#pragma unroll
            for (int k = 0; k < 32; ++k)
                g += s_h[k] * Whh[tid * 32 + k];
            s_g[tid] = g;
        }
        __syncthreads();
        if (tid < 32) {
            const float gi = s_g[tid];
            const float gf = s_g[32 + tid];
            const float gg = s_g[64 + tid];
            const float go = s_g[96 + tid];
            const float c  = sigm(gf) * s_c[tid] + sigm(gi) * tanhf(gg);
            const float h  = sigm(go) * tanhf(c);
            s_c[tid] = c;
            s_h[tid] = h;
            s_ys[t * 32 + tid] = h;
        }
        __syncthreads();
    }

    // ---- numeric MLP: 6 -> 64 -> 64, relu ---------------------------------
    if (tid < 64) {
        float acc = bn1[tid];
#pragma unroll
        for (int k = 0; k < 6; ++k) acc += num[k] * wn1[tid * 6 + k];
        s_y1[tid] = fmaxf(acc, 0.0f);
    }
    __syncthreads();
    if (tid < 64) {
        float acc = bn2[tid];
#pragma unroll
        for (int k = 0; k < 64; ++k) acc += s_y1[k] * wn2[tid * 64 + k];
        s_y2[tid] = fmaxf(acc, 0.0f);
    }
    __syncthreads();

    // ---- head + state write-out -------------------------------------------
    // d_out layout: out1[0:3] out2[3:6] h1[6:38] c1[38:70] h2[70:102] c2[102:134]
    float* out = A.out;
    if (tid < 3) {
        float acc = bo[tid];
        for (int k = 0; k < 128; ++k) acc += s_ys[k] * wo[tid * 192 + k];
#pragma unroll
        for (int k = 0; k < 64; ++k)  acc += s_y2[k] * wo[tid * 192 + 128 + k];
        out[3 * br + tid] = tanhf(acc);
    }
    if (tid < 32) {
        out[6 + 64 * br + tid]      = s_h[tid];
        out[6 + 64 * br + 32 + tid] = s_c[tid];
    }
}

extern "C" void kernel_launch(void* const* d_in, const int* in_sizes, int n_in,
                              void* d_out, int out_size, void* d_ws, size_t ws_size,
                              hipStream_t stream) {
    (void)in_sizes; (void)n_in; (void)out_size; (void)d_ws; (void)ws_size;
    KArgs A;
    for (int i = 0; i < 38; ++i) A.in[i] = (const float*)d_in[i];
    A.out = (float*)d_out;
    critic_branch_kernel<<<2, 256, 0, stream>>>(A);
}
